// QuantumNeuralLayer_7352984011340
// MI455X (gfx1250) — compile-verified, matrix operand order checked by reading, not run on hardware
//
#include <hip/hip_runtime.h>
#include <hip/hip_bf16.h>

// ---------------- problem constants ----------------
#define BATCH 256
#define FEAT  64
#define NQ    16
#define NL    2
#define HENC  128
#define HDEC  256
#define NST   65536          // 2^16 amplitudes
#define PI_F  3.14159265358979323846f
#define PROB_SCALE   65536.0f
#define PROB_UNSCALE (1.0f / 65536.0f)

// ---------------- WMMA vector types ----------------
typedef __attribute__((ext_vector_type(16))) _Float16 v16h;
typedef __attribute__((ext_vector_type(8)))  _Float16 v8h;
typedef __attribute__((ext_vector_type(8)))  float    v8f;

// ---------------- complex helpers ----------------
__device__ __forceinline__ float2 cmul(float2 a, float2 b) {
    return make_float2(a.x * b.x - a.y * b.y, a.x * b.y + a.y * b.x);
}
__device__ __forceinline__ float2 cadd(float2 a, float2 b) {
    return make_float2(a.x + b.x, a.y + b.y);
}

// Inverse of the CNOT-ring permutation (qubit q lives at bit 15-q).
// Forward ring: CNOT(q,q+1) for q=0..14 then CNOT(15,0) telescopes to
// b'_p = suffix-xor(b_p..b_15) for p<=14, b'_15 = xor(b_0..b_14).
// Inverse: r = i ^ (i>>1), then fix bits 15 and 14 with b'_0.
__device__ __forceinline__ unsigned invperm(unsigned i) {
    unsigned b0 = i & 1u;
    unsigned r = (i ^ (i >> 1)) & 0xFFFFu;
    r ^= (b0 << 15) | (b0 << 14);
    return r;
}

// ============================================================
// Kernel 1: classical encoder -> angles -> U3 2x2 complex gates
// One block per batch element, 128 threads.
// U layout in ws: per (b,l,q): 8 floats = U00.re,im, U01, U10, U11
// ============================================================
__global__ void encoder_kernel(const float* __restrict__ x,
                               const float* __restrict__ W1,
                               const float* __restrict__ b1,
                               const float* __restrict__ W2,
                               const float* __restrict__ b2,
                               const float* __restrict__ scaling,
                               float* __restrict__ Uws) {
    const int b = blockIdx.x;
    const int t = threadIdx.x;
    __shared__ float h[HENC];
    __shared__ float ang[NL * NQ * 3];

    // h = relu(x @ W1 + b1)
    {
        float acc = b1[t];
        const float* xr = x + b * FEAT;
        #pragma unroll 4
        for (int k = 0; k < FEAT; ++k) acc += xr[k] * W1[k * HENC + t];
        h[t] = acc > 0.f ? acc : 0.f;
    }
    __syncthreads();

    // p = tanh(h @ W2 + b2) * scaling[q] * pi
    if (t < NL * NQ * 3) {
        float acc = b2[t];
        #pragma unroll 4
        for (int k = 0; k < HENC; ++k) acc += h[k] * W2[k * (NL * NQ * 3) + t];
        int q = (t / 3) & (NQ - 1);
        ang[t] = tanhf(acc) * scaling[q] * PI_F;
    }
    __syncthreads();

    // U = Rz(lam) @ Ry(phi) @ Rx(theta)
    if (t < NL * NQ) {
        float th = ang[t * 3 + 0], ph = ang[t * 3 + 1], lm = ang[t * 3 + 2];
        float ct = cosf(th * 0.5f), st = sinf(th * 0.5f);
        float cp = cosf(ph * 0.5f), sp = sinf(ph * 0.5f);
        float cl = cosf(lm * 0.5f), sl = sinf(lm * 0.5f);
        // M = Ry@Rx
        float2 m00 = make_float2( cp * ct,  sp * st);
        float2 m01 = make_float2(-sp * ct, -cp * st);
        float2 m10 = make_float2( sp * ct, -cp * st);
        float2 m11 = make_float2( cp * ct, -sp * st);
        float2 em = make_float2(cl, -sl);   // e^{-i lam/2}
        float2 ep = make_float2(cl,  sl);   // e^{+i lam/2}
        float2 U00 = cmul(em, m00), U01 = cmul(em, m01);
        float2 U10 = cmul(ep, m10), U11 = cmul(ep, m11);
        float* dst = Uws + ((size_t)(b * NL) + (t >> 4)) * NQ * 8 + (t & 15) * 8;
        dst[0] = U00.x; dst[1] = U00.y; dst[2] = U01.x; dst[3] = U01.y;
        dst[4] = U10.x; dst[5] = U10.y; dst[6] = U11.x; dst[7] = U11.y;
    }
}

// ============================================================
// Kernel 2: fused  (layer-0 tensor-product init) + (CNOT-ring
// inverse-permute gather) + (layer-1 gates on bit positions 0..3).
// One thread owns 16 amplitudes. grid = (16, BATCH), block = 256.
// ============================================================
__device__ __forceinline__ void apply4(float2* v, const float2 (*U)[4]) {
    #pragma unroll
    for (int qb = 0; qb < 4; ++qb) {
        const int m = 1 << qb;
        #pragma unroll
        for (int j0 = 0; j0 < 16; ++j0) {
            if (j0 & m) continue;
            float2 a0 = v[j0], a1 = v[j0 | m];
            v[j0]     = cadd(cmul(U[qb][0], a0), cmul(U[qb][1], a1));
            v[j0 | m] = cadd(cmul(U[qb][2], a0), cmul(U[qb][3], a1));
        }
    }
}

__global__ void pass_init_kernel(const float* __restrict__ Uws,
                                 float2* __restrict__ state) {
    const int b = blockIdx.y;
    const int g = blockIdx.x * blockDim.x + threadIdx.x;   // 0..4095
    const int t = threadIdx.x;

    // col[p][c] : qubit 15-p acting on |0>, component c (U00 or U10)
    __shared__ float2 col[NQ][2];
    __shared__ float2 U2[4][4];
    const float2* F2 = (const float2*)Uws;   // 4 float2 per gate
    if (t < 32) {
        int p = t >> 1, c = t & 1;
        col[p][c] = F2[((size_t)(b * NL + 0) * NQ + (15 - p)) * 4 + c * 2];
    } else if (t < 48) {
        int qb = (t - 32) >> 2, e = (t - 32) & 3;
        U2[qb][e] = F2[((size_t)(b * NL + 1) * NQ + (15 - qb)) * 4 + e];
    }
    __syncthreads();

    float2 v[16];
    #pragma unroll
    for (int j = 0; j < 16; ++j) {
        unsigned idx = (unsigned)(g << 4) | (unsigned)j;   // bit positions 0..3 free
        unsigned src = invperm(idx);                       // pre-CNOT index
        float2 amp = make_float2(1.f, 0.f);
        #pragma unroll
        for (int p = 0; p < NQ; ++p) amp = cmul(amp, col[p][(src >> p) & 1u]);
        v[j] = amp;
    }
    apply4(v, U2);
    float2* st = state + (size_t)b * NST + (g << 4);
    #pragma unroll
    for (int j = 0; j < 16; ++j) st[j] = v[j];
}

// ============================================================
// Kernel 3: generic 4-qubit gate pass (layer 1, bit positions p0..p0+3)
// ============================================================
__global__ void gate_pass_kernel(const float* __restrict__ Uws,
                                 float2* __restrict__ state, int p0) {
    const int b = blockIdx.y;
    const int g = blockIdx.x * blockDim.x + threadIdx.x;   // 0..4095
    const int t = threadIdx.x;

    __shared__ float2 U2[4][4];
    const float2* F2 = (const float2*)Uws;
    if (t < 16) {
        int qb = t >> 2, e = t & 3;
        U2[qb][e] = F2[((size_t)(b * NL + 1) * NQ + (15 - (p0 + qb))) * 4 + e];
    }
    __syncthreads();

    const int low  = g & ((1 << p0) - 1);
    const int high = g >> p0;
    const int base = (high << (p0 + 4)) | low;
    float2* st = state + (size_t)b * NST;

    float2 v[16];
    #pragma unroll
    for (int j = 0; j < 16; ++j) v[j] = st[base | (j << p0)];
    apply4(v, U2);
    #pragma unroll
    for (int j = 0; j < 16; ++j) st[base | (j << p0)] = v[j];
}

// ============================================================
// Kernel 4: probs = |amp|^2 * 2^16 -> f16 (WMMA A operand).
// Power-of-two prescale keeps tiny probabilities in f16 normal range;
// GEMM1 epilogue multiplies by 2^-16 (exact).
// ============================================================
__global__ void probs_kernel(const float2* __restrict__ st,
                             _Float16* __restrict__ probs) {
    const int i = blockIdx.x * blockDim.x + threadIdx.x;
    float2 a = st[i];
    probs[i] = (_Float16)((a.x * a.x + a.y * a.y) * PROB_SCALE);
}

// ============================================================
// Kernel 5: transpose+convert f32 [K,N] -> f16 [N,K] (for WMMA B)
// ============================================================
__global__ void transpose_f32_to_f16(const float* __restrict__ in,
                                     _Float16* __restrict__ out,
                                     int K, int N) {
    __shared__ float tile[16][17];
    const int k0 = blockIdx.x * 16, n0 = blockIdx.y * 16;
    const int tx = threadIdx.x, ty = threadIdx.y;
    tile[ty][tx] = in[(size_t)(k0 + ty) * N + (n0 + tx)];
    __syncthreads();
    out[(size_t)(n0 + ty) * K + k0 + tx] = (_Float16)tile[tx][ty];
}

// ============================================================
// WMMA fragment load from a per-lane base pointer (already offset by
// row*ld + (lane>>4)*8): lane L<16 -> row L, K {0..7, 16..23};
// lane L+16 -> row L, K {8..15, 24..31}.
// ============================================================
__device__ __forceinline__ v16h load_frag(const _Float16* __restrict__ p, int kb) {
    union { v16h v; v8h h[2]; } u;
    u.h[0] = *(const v8h*)(p + kb);
    u.h[1] = *(const v8h*)(p + kb + 16);
    return u.v;
}

// ============================================================
// Kernel 6: GEMM1  h1 = relu(probs @ Wd1 * 2^-16 + bd1) -> f16
// [BATCH, HDEC].  Block = 8 waves, split-K (K/8 per wave, SGPR
// bounds via readfirstlane).  Explicit ping-pong double buffering:
// the loads after each WMMA refill the buffer that WMMA just
// consumed (used again two WMMAs later) -> no rotation copies.
// LDS split-K reduction; wave 0 does the epilogue.
// ============================================================
__global__ void gemm1_wmma_kernel(const _Float16* __restrict__ A,
                                  const _Float16* __restrict__ Bt,
                                  const float* __restrict__ bias,
                                  _Float16* __restrict__ C,
                                  int K, int N) {
    const int tid  = threadIdx.x;
    const int lane = tid & 31;
    // wave index is wave-uniform: force it into an SGPR so the k-loop
    // bounds are scalar (keeps EXEC all-ones around WMMA).
    const int wave = __builtin_amdgcn_readfirstlane(tid >> 5);   // 0..7
    const int tm = blockIdx.x * 16, tn = blockIdx.y * 16;
    const int rowA = tm + (lane & 15);
    const int rowB = tn + (lane & 15);
    const int kSlice = K >> 3;                 // 8192 (256 chunks of 32)
    const int k0   = wave * kSlice;            // scalar
    const int kEnd = k0 + kSlice;              // scalar

    const _Float16* baseA = A  + (size_t)rowA * K + ((lane >> 4) << 3);
    const _Float16* baseB = Bt + (size_t)rowB * K + ((lane >> 4) << 3);

    v8f c = {};
    v16h a0 = load_frag(baseA, k0);
    v16h b0 = load_frag(baseB, k0);
    v16h a1 = load_frag(baseA, k0 + 32);
    v16h b1 = load_frag(baseB, k0 + 32);
    int kb = k0;
    for (; kb + 64 < kEnd; kb += 64) {
        c = __builtin_amdgcn_wmma_f32_16x16x32_f16(false, a0, false, b0,
                                                   (short)0, c, false, false);
        a0 = load_frag(baseA, kb + 64);
        b0 = load_frag(baseB, kb + 64);
        __builtin_prefetch(baseA + kb + 256, 0, 3);  // speculative, OK past kEnd
        c = __builtin_amdgcn_wmma_f32_16x16x32_f16(false, a1, false, b1,
                                                   (short)0, c, false, false);
        a1 = load_frag(baseA, kb + 96);
        b1 = load_frag(baseB, kb + 96);
        __builtin_prefetch(baseB + kb + 256, 0, 3);
    }
    c = __builtin_amdgcn_wmma_f32_16x16x32_f16(false, a0, false, b0,
                                               (short)0, c, false, false);
    c = __builtin_amdgcn_wmma_f32_16x16x32_f16(false, a1, false, b1,
                                               (short)0, c, false, false);

    // cross-wave split-K reduction (stride 9 -> conflict-free banks)
    __shared__ float red[8 * 32 * 9];
    #pragma unroll
    for (int r = 0; r < 8; ++r) red[(wave * 32 + lane) * 9 + r] = c[r];
    __syncthreads();

    if (wave == 0) {
        float acc[8];
        #pragma unroll
        for (int r = 0; r < 8; ++r) {
            float s = 0.f;
            #pragma unroll
            for (int w = 0; w < 8; ++w) s += red[(w * 32 + lane) * 9 + r];
            acc[r] = s;
        }
        const int col = tn + (lane & 15);
        const float bv = bias[col];
        const int rbase = tm + ((lane >> 4) << 3);
        #pragma unroll
        for (int r = 0; r < 8; ++r) {
            float val = acc[r] * PROB_UNSCALE + bv;
            val = val > 0.f ? val : 0.f;
            C[(size_t)(rbase + r) * N + col] = (_Float16)val;
        }
    }
}

// ============================================================
// Kernel 7: GEMM2  out = relu(h1 @ Wd2 + bd2) -> f32 [BATCH, FEAT]
// One wave per tile, K = 256 (8 WMMA), ping-pong pipeline.
// ============================================================
__global__ void gemm2_wmma_kernel(const _Float16* __restrict__ A,
                                  const _Float16* __restrict__ Bt,
                                  const float* __restrict__ bias,
                                  float* __restrict__ C,
                                  int K, int N) {
    const int lane = threadIdx.x;
    const int tm = blockIdx.x * 16, tn = blockIdx.y * 16;
    const int rowA = tm + (lane & 15);
    const int rowB = tn + (lane & 15);
    const _Float16* baseA = A  + (size_t)rowA * K + ((lane >> 4) << 3);
    const _Float16* baseB = Bt + (size_t)rowB * K + ((lane >> 4) << 3);

    v8f c = {};
    v16h a0 = load_frag(baseA, 0);
    v16h b0 = load_frag(baseB, 0);
    v16h a1 = load_frag(baseA, 32);
    v16h b1 = load_frag(baseB, 32);
    int kb = 0;
    for (; kb + 64 < K; kb += 64) {
        c = __builtin_amdgcn_wmma_f32_16x16x32_f16(false, a0, false, b0,
                                                   (short)0, c, false, false);
        a0 = load_frag(baseA, kb + 64);
        b0 = load_frag(baseB, kb + 64);
        c = __builtin_amdgcn_wmma_f32_16x16x32_f16(false, a1, false, b1,
                                                   (short)0, c, false, false);
        a1 = load_frag(baseA, kb + 96);
        b1 = load_frag(baseB, kb + 96);
    }
    c = __builtin_amdgcn_wmma_f32_16x16x32_f16(false, a0, false, b0,
                                               (short)0, c, false, false);
    c = __builtin_amdgcn_wmma_f32_16x16x32_f16(false, a1, false, b1,
                                               (short)0, c, false, false);

    const int col = tn + (lane & 15);
    const float bv = bias[col];
    const int rbase = tm + ((lane >> 4) << 3);
    #pragma unroll
    for (int r = 0; r < 8; ++r) {
        float val = c[r] + bv;
        val = val > 0.f ? val : 0.f;
        C[(size_t)(rbase + r) * N + col] = val;
    }
}

// ============================================================
// launch
// ============================================================
extern "C" void kernel_launch(void* const* d_in, const int* in_sizes, int n_in,
                              void* d_out, int out_size, void* d_ws, size_t ws_size,
                              hipStream_t stream) {
    const float* x       = (const float*)d_in[0];
    const float* W1      = (const float*)d_in[1];
    const float* b1      = (const float*)d_in[2];
    const float* W2      = (const float*)d_in[3];
    const float* b2      = (const float*)d_in[4];
    const float* scaling = (const float*)d_in[5];
    const float* Wd1     = (const float*)d_in[6];
    const float* bd1     = (const float*)d_in[7];
    const float* Wd2     = (const float*)d_in[8];
    const float* bd2     = (const float*)d_in[9];
    float* out = (float*)d_out;

    // workspace carve-up (all offsets 256B-aligned)
    char* ws = (char*)d_ws;
    float*     Uws   = (float*)ws;                                 // 256 KB
    float2*    state = (float2*)(ws + (size_t)262144);             // 128 MB
    _Float16*  probs = (_Float16*)(ws + (size_t)262144 + 134217728);      // 32 MB
    _Float16*  Wd1T  = probs + (size_t)BATCH * NST;                // 32 MB
    _Float16*  h1    = Wd1T  + (size_t)HDEC * NST;                 // 128 KB
    _Float16*  Wd2T  = h1    + (size_t)BATCH * HDEC;               // 32 KB

    // 1. encoder -> U3 gate matrices
    encoder_kernel<<<BATCH, HENC, 0, stream>>>(x, W1, b1, W2, b2, scaling, Uws);

    // 2. weight transpose+convert (f32 [K,N] -> f16 [N,K])
    transpose_f32_to_f16<<<dim3(NST / 16, HDEC / 16), dim3(16, 16), 0, stream>>>(
        Wd1, Wd1T, NST, HDEC);
    transpose_f32_to_f16<<<dim3(HDEC / 16, FEAT / 16), dim3(16, 16), 0, stream>>>(
        Wd2, Wd2T, HDEC, FEAT);

    // 3. quantum sim: tensor-product init + CNOT permute + layer-1 gates 0..3
    pass_init_kernel<<<dim3(16, BATCH), 256, 0, stream>>>(Uws, state);
    // remaining layer-1 gates, 4 bit-positions per pass
    gate_pass_kernel<<<dim3(16, BATCH), 256, 0, stream>>>(Uws, state, 4);
    gate_pass_kernel<<<dim3(16, BATCH), 256, 0, stream>>>(Uws, state, 8);
    gate_pass_kernel<<<dim3(16, BATCH), 256, 0, stream>>>(Uws, state, 12);

    // 4. probabilities -> f16 (x 2^16)
    probs_kernel<<<(BATCH * NST) / 256, 256, 0, stream>>>(state, probs);

    // 5. decoder GEMMs on WMMA
    gemm1_wmma_kernel<<<dim3(BATCH / 16, HDEC / 16), 256, 0, stream>>>(
        probs, Wd1T, bd1, h1, NST, HDEC);
    gemm2_wmma_kernel<<<dim3(BATCH / 16, FEAT / 16), 32, 0, stream>>>(
        h1, Wd2T, bd2, out, HDEC, FEAT);
}
